// MRFLoss_9414568313627
// MI455X (gfx1250) — compile-verified, matrix operand order checked by reading, not run on hardware
//
#include <hip/hip_runtime.h>

typedef __attribute__((ext_vector_type(16))) _Float16 v16h;
typedef __attribute__((ext_vector_type(8)))  _Float16 v8h;
typedef __attribute__((ext_vector_type(8)))  float    v8f;

#define NT     2            // 16-col tiles per wave iteration (N blocking)
#define NBLK   (NT * 16)    // columns consumed per wave iteration
#define NWAVES 8            // 256 threads / wave32
#define NEG_BIG (-3.4e38f)

// ---------------------------------------------------------------------------
// Prep: per patch, compute L2 norm over K=C*9 window elements, write the
// normalized patch row in f16 (row-major [Lp x Kp]); pad rows are zeroed.
// One wave per patch.
// ---------------------------------------------------------------------------
__global__ void mrf_prep_kernel(const float* __restrict__ F, _Float16* __restrict__ P,
                                int C, int H, int W, int K, int Kp, int Li, int Lp)
{
  const int p    = blockIdx.x * NWAVES + (threadIdx.x >> 5);
  const int lane = threadIdx.x & 31;
  if (p >= Lp) return;
  _Float16* row = P + (size_t)p * Kp;
  if (p >= Li) {                       // zero padding row
    for (int k = lane; k < Kp; k += 32) row[k] = (_Float16)0.f;
    return;
  }
  const int w = W - 2;
  const int y = p / w, x = p - y * w;
  const float* base = F + (size_t)y * W + x;
  const int HW = H * W;

  float ss = 0.f;
  for (int k = lane; k < K; k += 32) {
    const int c = k / 9, pos = k - c * 9;
    const int i = pos / 3, j = pos - i * 3;
    const float v = base[(size_t)c * HW + i * W + j];
    ss += v * v;
  }
  #pragma unroll
  for (int m = 16; m >= 1; m >>= 1) ss += __shfl_xor(ss, m, 32);
  const float inv = rsqrtf(ss);

  for (int k = lane; k < Kp; k += 32) {
    float v = 0.f;
    if (k < K) {
      const int c = k / 9, pos = k - c * 9;
      const int i = pos / 3, j = pos - i * 3;
      v = base[(size_t)c * HW + i * W + j] * inv;
    }
    row[k] = (_Float16)v;
  }
}

// ---------------------------------------------------------------------------
// Fused GEMM + row-argmax:  sim = A (Li x K) * B^T (Ls x K); nn_idx[row] =
// argmax_col sim[row][col].  A block (MT*16 rows x K) staged in LDS via CDNA5
// async global->LDS DMA; 8 waves split column tile-pairs; each wave keeps
// MT x NT accumulator tiles (each LDS A-fragment feeds NT WMMAs) plus a
// running per-(row,lane) best reduced by wave shuffles + LDS at the end.
// MT=4 for K=2304 (288KB LDS block), MT=2 for K=4608 (288KB LDS block).
// ---------------------------------------------------------------------------
template <int MT>
__global__ void mrf_gemm_argmax_kernel(const _Float16* __restrict__ A,
                                       const _Float16* __restrict__ B,
                                       int* __restrict__ nn_idx,
                                       int K, int Li, int Ls, int nPairs)
{
  constexpr int MBLK = MT * 16;
  extern __shared__ _Float16 sA[];           // [MBLK][K] + merge buffers
  const int tid   = threadIdx.x;
  const int lane  = tid & 31;
  const int wave  = tid >> 5;                // 0..7
  const int half  = lane >> 4;               // 0/1
  const int l15   = lane & 15;
  const int mBase = blockIdx.x * MBLK;

  float* mergeV = (float*)(sA + (size_t)MBLK * K);
  int*   mergeI = (int*)(mergeV + NWAVES * MBLK);

  // --- async DMA stage of A block into LDS (ASYNCcnt path, no VGPR bounce) ---
  {
    const char* src = (const char*)(A + (size_t)mBase * K);
    const unsigned dstBase = (unsigned)(uintptr_t)sA;   // low 32b of shared aperture = LDS offset
    const int total = MBLK * K * 2;                     // bytes; multiple of 256*16
    for (int b = tid * 16; b < total; b += 256 * 16) {
      const unsigned d = dstBase + (unsigned)b;
      const char*    s = src + b;
      asm volatile("global_load_async_to_lds_b128 %0, %1, off"
                   :: "v"(d), "v"(s) : "memory");
    }
    asm volatile("s_wait_asynccnt 0" ::: "memory");
  }
  __syncthreads();

  float bestV[MT][8];
  int   bestI[MT][8];
  #pragma unroll
  for (int t = 0; t < MT; ++t)
    #pragma unroll
    for (int i = 0; i < 8; ++i) { bestV[t][i] = NEG_BIG; bestI[t][i] = 0; }

  for (int np = wave; np < nPairs; np += NWAVES) {     // wave-uniform loop
    const int col0 = np * NBLK + l15;                  // lane's column, tile 0
    const int col1 = col0 + 16;                        // lane's column, tile 1
    const _Float16* brow0 = B + (size_t)col0 * K;      // style patch rows (K-major)
    const _Float16* brow1 = B + (size_t)col1 * K;

    v8f c[MT][NT];
    #pragma unroll
    for (int t = 0; t < MT; ++t)
      #pragma unroll
      for (int u = 0; u < NT; ++u) c[t][u] = (v8f){};

    for (int k0 = 0; k0 < K; k0 += 32) {
      __builtin_prefetch(brow0 + k0 + 512, 0, 3);      // global_prefetch_b8 (near)
      __builtin_prefetch(brow1 + k0 + 512, 0, 3);
      // B fragments: per ISA layout lane-half h holds K in {8h..8h+7} u {16+8h..16+8h+7}
      v8h b0lo = *(const v8h*)(brow0 + k0 + 8 * half);
      v8h b0hi = *(const v8h*)(brow0 + k0 + 16 + 8 * half);
      v16h bf0 = __builtin_shufflevector(b0lo, b0hi, 0,1,2,3,4,5,6,7,8,9,10,11,12,13,14,15);
      v8h b1lo = *(const v8h*)(brow1 + k0 + 8 * half);
      v8h b1hi = *(const v8h*)(brow1 + k0 + 16 + 8 * half);
      v16h bf1 = __builtin_shufflevector(b1lo, b1hi, 0,1,2,3,4,5,6,7,8,9,10,11,12,13,14,15);
      #pragma unroll
      for (int t = 0; t < MT; ++t) {
        const _Float16* arow = sA + (size_t)(t * 16 + l15) * K;
        v8h alo = *(const v8h*)(arow + k0 + 8 * half);
        v8h ahi = *(const v8h*)(arow + k0 + 16 + 8 * half);
        v16h af = __builtin_shufflevector(alo, ahi, 0,1,2,3,4,5,6,7,8,9,10,11,12,13,14,15);
        // one LDS fragment feeds NT WMMAs -> 1 ds_load_b128 per WMMA
        c[t][0] = __builtin_amdgcn_wmma_f32_16x16x32_f16(
                      false, af, false, bf0, (short)0, c[t][0], false, false);
        c[t][1] = __builtin_amdgcn_wmma_f32_16x16x32_f16(
                      false, af, false, bf1, (short)0, c[t][1], false, false);
      }
    }

    // fold tiles into running per-lane best (mask padded columns)
    #pragma unroll
    for (int u = 0; u < NT; ++u) {
      const int col = (u == 0) ? col0 : col1;
      const bool ok = col < Ls;
      #pragma unroll
      for (int t = 0; t < MT; ++t)
        #pragma unroll
        for (int i = 0; i < 8; ++i) {
          const float v = ok ? c[t][u][i] : NEG_BIG;
          if (v > bestV[t][i] || (v == bestV[t][i] && col < bestI[t][i])) {
            bestV[t][i] = v; bestI[t][i] = col;
          }
        }
    }
  }

  // --- reduce across the 16 lanes of each half (C-layout: lane = column) ---
  #pragma unroll
  for (int t = 0; t < MT; ++t)
    #pragma unroll
    for (int i = 0; i < 8; ++i) {
      float v = bestV[t][i]; int ci = bestI[t][i];
      #pragma unroll
      for (int m = 1; m < 16; m <<= 1) {
        const float v2 = __shfl_xor(v, m, 32);
        const int   c2 = __shfl_xor(ci, m, 32);
        if (v2 > v || (v2 == v && c2 < ci)) { v = v2; ci = c2; }
      }
      bestV[t][i] = v; bestI[t][i] = ci;
    }

  if (l15 == 0) {
    #pragma unroll
    for (int t = 0; t < MT; ++t)
      #pragma unroll
      for (int i = 0; i < 8; ++i) {
        const int r = t * 16 + 8 * half + i;           // row within block
        mergeV[wave * MBLK + r] = bestV[t][i];
        mergeI[wave * MBLK + r] = bestI[t][i];
      }
  }
  __syncthreads();

  if (tid < MBLK) {                                    // cross-wave merge
    float v = mergeV[tid]; int ci = mergeI[tid];
    for (int w2 = 1; w2 < NWAVES; ++w2) {
      const float v2 = mergeV[w2 * MBLK + tid];
      const int   c2 = mergeI[w2 * MBLK + tid];
      if (v2 > v || (v2 == v && c2 < ci)) { v = v2; ci = c2; }
    }
    const int gr = mBase + tid;
    if (gr < Li) nn_idx[gr] = ci;
  }
}

// ---------------------------------------------------------------------------
// Loss: sum_k<rows (ip[k,l] - sp[k,nn[l]])^2, raw (unnormalized) patches
// recomputed directly from the f32 feature maps. One wave per input patch.
// ---------------------------------------------------------------------------
__global__ void mrf_loss_kernel(const float* __restrict__ Fin, const float* __restrict__ Fst,
                                const int* __restrict__ nn,
                                int H, int W, int rows, int Li, float* __restrict__ acc)
{
  const int p    = blockIdx.x * NWAVES + (threadIdx.x >> 5);
  const int lane = threadIdx.x & 31;
  if (p >= Li) return;
  const int w = W - 2;
  const int yi = p / w, xi = p - yi * w;
  const int q  = nn[p];
  const int ys = q / w, xs = q - ys * w;
  const float* bi = Fin + (size_t)yi * W + xi;
  const float* bs = Fst + (size_t)ys * W + xs;
  const int HW = H * W;

  float s = 0.f;
  for (int k = lane; k < rows; k += 32) {
    const int c = k / 9, pos = k - c * 9;
    const int i = pos / 3, j = pos - i * 3;
    const size_t off = (size_t)c * HW + i * W + j;
    const float d = bi[off] - bs[off];
    s += d * d;
  }
  #pragma unroll
  for (int m = 16; m >= 1; m >>= 1) s += __shfl_xor(s, m, 32);
  if (lane == 0) atomicAdd(acc, s);
}

__global__ void mrf_zero_kernel(float* acc) { if (threadIdx.x == 0) *acc = 0.f; }

__global__ void mrf_finalize_kernel(const float* __restrict__ acc, float* __restrict__ out)
{
  if (threadIdx.x == 0) {
    const float s = *acc;
    out[0] = s;      // total = 1*0 + 1*style + 1*0
    out[1] = 0.f;    // content
    out[2] = s;      // style
    out[3] = 0.f;    // tv
  }
}

// ---------------------------------------------------------------------------
extern "C" void kernel_launch(void* const* d_in, const int* in_sizes, int n_in,
                              void* d_out, int out_size, void* d_ws, size_t ws_size,
                              hipStream_t stream) {
  (void)in_sizes; (void)n_in; (void)out_size; (void)ws_size;
  const float* sf0 = (const float*)d_in[0];   // style_f0 [1,256,96,96]
  const float* if0 = (const float*)d_in[1];   // input_f0
  const float* sf1 = (const float*)d_in[2];   // style_f1 [1,512,48,48]
  const float* if1 = (const float*)d_in[3];   // input_f1
  float* out = (float*)d_out;

  const int C0 = 256, H0 = 96, W0 = 96;
  const int C1 = 512, H1 = 48, W1 = 48;
  const int K0 = C0 * 9, K1 = C1 * 9;                       // 2304, 4608 (mult of 32)
  const int L0 = (H0 - 2) * (W0 - 2);                       // 8836
  const int L1 = (H1 - 2) * (W1 - 2);                       // 2116
  const int MB0 = 4 * 16, MB1 = 2 * 16;                     // per-scale M blocks
  const int Lp0  = ((L0 + MB0 - 1) / MB0) * MB0;            // A-row padding
  const int Lp1  = ((L1 + MB1 - 1) / MB1) * MB1;
  const int nP0  = (L0 + NBLK - 1) / NBLK;                  // column tile-pairs
  const int nP1  = (L1 + NBLK - 1) / NBLK;
  const int Lsp0 = nP0 * NBLK, Lsp1 = nP1 * NBLK;           // B-row padding (32)
  const int rows0 = (C0 - 3) * 9, rows1 = (C1 - 3) * 9;

  // workspace carve-out (256B aligned)
  char* ws = (char*)d_ws;
  size_t off = 0;
  auto carve = [&](size_t bytes) -> char* {
    char* p = ws + off;
    off = (off + bytes + 255) & ~(size_t)255;
    return p;
  };
  float*    acc = (float*)carve(sizeof(float));
  int*      nn0 = (int*)carve((size_t)L0 * sizeof(int));
  int*      nn1 = (int*)carve((size_t)L1 * sizeof(int));
  _Float16* A0  = (_Float16*)carve((size_t)Lp0  * K0 * 2);  // input patches, normalized
  _Float16* B0  = (_Float16*)carve((size_t)Lsp0 * K0 * 2);  // style patches, normalized
  _Float16* A1  = (_Float16*)carve((size_t)Lp1  * K1 * 2);
  _Float16* B1  = (_Float16*)carve((size_t)Lsp1 * K1 * 2);

  const size_t smem0 = (size_t)MB0 * K0 * 2 + (size_t)NWAVES * MB0 * 8;  // ~292 KB
  const size_t smem1 = (size_t)MB1 * K1 * 2 + (size_t)NWAVES * MB1 * 8;  // ~290 KB
  (void)hipFuncSetAttribute((const void*)mrf_gemm_argmax_kernel<4>,
                            hipFuncAttributeMaxDynamicSharedMemorySize, (int)smem0);
  (void)hipFuncSetAttribute((const void*)mrf_gemm_argmax_kernel<2>,
                            hipFuncAttributeMaxDynamicSharedMemorySize, (int)smem1);

  mrf_zero_kernel<<<1, 32, 0, stream>>>(acc);

  // normalize + pack f16 patch matrices
  mrf_prep_kernel<<<(Lp0  + NWAVES - 1) / NWAVES, 256, 0, stream>>>(if0, A0, C0, H0, W0, K0, K0, L0, Lp0);
  mrf_prep_kernel<<<(Lsp0 + NWAVES - 1) / NWAVES, 256, 0, stream>>>(sf0, B0, C0, H0, W0, K0, K0, L0, Lsp0);
  mrf_prep_kernel<<<(Lp1  + NWAVES - 1) / NWAVES, 256, 0, stream>>>(if1, A1, C1, H1, W1, K1, K1, L1, Lp1);
  mrf_prep_kernel<<<(Lsp1 + NWAVES - 1) / NWAVES, 256, 0, stream>>>(sf1, B1, C1, H1, W1, K1, K1, L1, Lsp1);

  // fused cosine-sim GEMM + argmax (WMMA f16 -> f32)
  mrf_gemm_argmax_kernel<4><<<Lp0 / MB0, 256, smem0, stream>>>(A0, B0, nn0, K0, L0, L0, nP0);
  mrf_gemm_argmax_kernel<2><<<Lp1 / MB1, 256, smem1, stream>>>(A1, B1, nn1, K1, L1, L1, nP1);

  // nearest-neighbor squared-diff over raw patches
  mrf_loss_kernel<<<(L0 + NWAVES - 1) / NWAVES, 256, 0, stream>>>(if0, sf0, nn0, H0, W0, rows0, L0, acc);
  mrf_loss_kernel<<<(L1 + NWAVES - 1) / NWAVES, 256, 0, stream>>>(if1, sf1, nn1, H1, W1, rows1, L1, acc);

  mrf_finalize_kernel<<<1, 32, 0, stream>>>(acc, out);
}
